// Model_53300544143399
// MI455X (gfx1250) — matrix operand order repaired and checked by reading, not compiled, not run on hardware
//
#include <hip/hip_runtime.h>

typedef _Float16 h16;
typedef __attribute__((ext_vector_type(16))) _Float16 v16h;
typedef __attribute__((ext_vector_type(8)))  _Float16 v8h;
typedef __attribute__((ext_vector_type(8)))  float    v8f;

#define TPB 256

// ---------------------------------------------------------------------------
// Block-wide max -> atomicMax on float bits (values are >= 0, bit compare ok)
// ---------------------------------------------------------------------------
__device__ inline void blockMaxAtomic(float v, float* slot) {
  __shared__ float red[TPB];
  int t = threadIdx.x;
  red[t] = v;
  __syncthreads();
  for (int s = TPB / 2; s > 0; s >>= 1) {
    if (t < s) red[t] = fmaxf(red[t], red[t + s]);
    __syncthreads();
  }
  if (t == 0) atomicMax((unsigned int*)slot, __float_as_uint(red[0]));
}

__global__ void k_init(float* slots, int n) {
  int i = threadIdx.x;
  if (i < n) ((unsigned int*)slots)[i] = 0u;
}

// per-output-channel abs-max over contiguous rows of length Kin
__global__ void k_amax_perchan(const float* __restrict__ w, int Kin,
                               float* __restrict__ amax) {
  int o = blockIdx.x;
  const float* row = w + (size_t)o * Kin;
  float m = 0.f;
  for (int i = threadIdx.x; i < Kin; i += TPB) m = fmaxf(m, fabsf(row[i]));
  __shared__ float red[TPB];
  red[threadIdx.x] = m;
  __syncthreads();
  for (int s = TPB / 2; s > 0; s >>= 1) {
    if (threadIdx.x < s) red[threadIdx.x] = fmaxf(red[threadIdx.x], red[threadIdx.x + s]);
    __syncthreads();
  }
  if (threadIdx.x == 0) amax[o] = red[0];
}

// reduce per-row amax to a single tensor-wide amax, broadcast back (fw2 path)
__global__ void k_amax_all(float* amax, int O) {
  __shared__ float red[TPB];
  float m = 0.f;
  for (int i = threadIdx.x; i < O; i += TPB) m = fmaxf(m, amax[i]);
  red[threadIdx.x] = m;
  __syncthreads();
  for (int s = TPB / 2; s > 0; s >>= 1) {
    if (threadIdx.x < s) red[threadIdx.x] = fmaxf(red[threadIdx.x], red[threadIdx.x + s]);
    __syncthreads();
  }
  float g = red[0];
  for (int i = threadIdx.x; i < O; i += TPB) amax[i] = g;
}

// conv1 weights: quantize in-place order (OIHW flat), kept f32 for scalar conv
__global__ void k_quant_conv1(const float* __restrict__ w, const float* __restrict__ amax,
                              const int* __restrict__ bwp, float* __restrict__ wq,
                              int total, int Kin) {
  int i = blockIdx.x * TPB + threadIdx.x;
  if (i >= total) return;
  int bw = bwp[0];
  float qm = (float)((1 << (bw - 1)) - 1);
  int o = i / Kin;
  float s = fmaxf(amax[o], 1e-8f) / qm;
  float q = rintf(w[i] / s);
  q = fminf(fmaxf(q, -qm), qm);
  wq[i] = q * s;
}

// Quantize weights (OIHW, per-channel scale) into WMMA-B fragment order:
// packed[((kt*Ntiles+nt)*32 + L)*16 + j] = B[kt*32 + (L/16)*8 + (j%8) + (j/8)*16][nt*16 + L%16]
// with GEMM k = (ky*KW + kx)*Cc + c  (matches NHWC im2col on the A side).
__global__ void k_pack_w(const float* __restrict__ w, const float* __restrict__ amax,
                         const int* __restrict__ bwp, h16* __restrict__ packed,
                         int O, int Cc, int KH, int KW, int Ktiles, int Ntiles) {
  int i = blockIdx.x * TPB + threadIdx.x;
  int total = Ktiles * Ntiles * 512;
  if (i >= total) return;
  int bw = bwp[0];
  float qm = (float)((1 << (bw - 1)) - 1);
  int tile = i >> 9, within = i & 511;
  int L = within >> 4, j = within & 15;
  int kt = tile / Ntiles, nt = tile % Ntiles;
  int k = kt * 32 + ((L >> 4) << 3) + (j & 7) + ((j >> 3) << 4);
  int n = nt * 16 + (L & 15);
  int Kin = Cc * KH * KW;
  float val = 0.f;
  if (k < Kin && n < O) {
    int c = k % Cc, t = k / Cc;
    int kx = t % KW, ky = t / KW;
    float s = fmaxf(amax[n], 1e-8f) / qm;
    float x = w[(((size_t)n * Cc + c) * KH + ky) * KW + kx];
    float q = rintf(x / s);
    q = fminf(fmaxf(q, -qm), qm);
    val = q * s;
  }
  packed[i] = (h16)val;
}

// conv1: NCHW f32 input, K=27 (too small for WMMA) -> scalar FMA.
// Fuses ReLU + global-max for the quant_relu scale. Output NHWC f16.
__global__ void k_conv1(const float* __restrict__ x, const float* __restrict__ wq,
                        h16* __restrict__ r1, float* __restrict__ maxslot, int Btot) {
  size_t total = (size_t)Btot * 30 * 30 * 64;
  size_t i = (size_t)blockIdx.x * TPB + threadIdx.x;
  float r = 0.f;
  if (i < total) {
    int oc = (int)(i & 63);
    size_t p = i >> 6;
    int ox = (int)(p % 30); p /= 30;
    int oy = (int)(p % 30);
    int b  = (int)(p / 30);
    const float* xb   = x + (size_t)b * 3 * 32 * 32;
    const float* wrow = wq + oc * 27;
    float sum = 0.f;
#pragma unroll
    for (int c = 0; c < 3; ++c)
#pragma unroll
      for (int ky = 0; ky < 3; ++ky)
#pragma unroll
        for (int kx = 0; kx < 3; ++kx)
          sum += xb[(c * 32 + oy + ky) * 32 + ox + kx] * wrow[c * 9 + ky * 3 + kx];
    r = fmaxf(sum, 0.f);
    r1[i] = (h16)r;
  }
  blockMaxAtomic(r, maxslot);
}

// quant_relu (scale from global max) + 2x2 maxpool, NHWC f16 -> NHWC f16.
// pool(quant(x)) == quant(pool(x)) since quant is monotone: max first, quant once.
__global__ void k_quant_pool(const h16* __restrict__ r, const float* __restrict__ maxslot,
                             const int* __restrict__ bwp, h16* __restrict__ out,
                             int Btot, int Ho, int Wo, int C) {
  size_t total = (size_t)Btot * Ho * Wo * C;
  size_t i = (size_t)blockIdx.x * TPB + threadIdx.x;
  if (i >= total) return;
  int bw = bwp[0];
  float qm = (float)((1 << bw) - 1);
  float s = fmaxf(maxslot[0], 1e-8f) / qm;
  int c = (int)(i % C);
  size_t p = i / C;
  int ox = (int)(p % Wo); p /= Wo;
  int oy = (int)(p % Ho);
  int b  = (int)(p / Ho);
  int Win = Wo * 2, Hin = Ho * 2;
  const h16* base = r + (((size_t)b * Hin + 2 * oy) * Win + 2 * ox) * C + c;
  float m = (float)base[0];
  m = fmaxf(m, (float)base[C]);
  m = fmaxf(m, (float)base[(size_t)Win * C]);
  m = fmaxf(m, (float)base[(size_t)Win * C + C]);
  float q = fminf(fmaxf(rintf(m / s), 0.f), qm) * s;
  out[i] = (h16)q;
}

// ---------------------------------------------------------------------------
// Implicit-GEMM conv / GEMM via v_wmma_f32_16x16x32_f16 (wave32).
// CC/KWC/STRIDE/KTILES/NTW are compile-time: the fully-unrolled K-loop
// constant-folds the k -> (ky,kx,c) decomposition (CC power of two, kt is an
// unroll constant, koff only has bit 3 unknown -> known-bits folds the rest).
// Each wave computes a 16 x (16*NTW) output tile: ONE shared A fragment feeds
// NTW WMMAs per K-step (convs: NTW=4 covers all N=64 -> 4x fewer A fetches,
// 4 wmma per 10 b128 loads). B is pre-packed in fragment order (one
// contiguous 32B load per lane per n-tile). Fuses ReLU + block max + atomicMax.
// ---------------------------------------------------------------------------
template <int CC, int KWC, int STRIDE, int KTILES, int NTW>
__global__ void k_gemm_wmma(const h16* __restrict__ in, const h16* __restrict__ pw,
                            h16* __restrict__ outH, float* __restrict__ outF,
                            float* __restrict__ maxslot,
                            int Hin, int Win,
                            int Ho, int Wo, int Ntiles,
                            int Mtiles, int Nout, int Nstride, int relu) {
  int wid  = (blockIdx.x * TPB + threadIdx.x) >> 5;
  int lane = threadIdx.x & 31;
  float wmax = 0.f;
  int ngroups = Ntiles / NTW;
  int totalW = Mtiles * ngroups;
  if (wid < totalW) {               // wave-uniform: EXEC stays all-1s for WMMA
    int nt0 = (wid % ngroups) * NTW;
    int mt  = wid / ngroups;
    int rowA = lane & 15;
    int koff = (lane >> 4) << 3;    // 0 or 8 (only bit 3 unknown)
    int m  = mt * 16 + rowA;        // output pixel (NHWC linear)
    int px = m % Wo;
    int rest = m / Wo;
    int py = rest % Ho;
    int b  = rest / Ho;

    // hoisted per-lane A base: pixel origin + koff channel offset
    const h16* abase = in + (((size_t)b * Hin + py * STRIDE) * Win + px * STRIDE) * CC + koff;
    const h16* bbase = pw + (((size_t)nt0 * 32) + lane) * 16;

    v8f acc[NTW];
#pragma unroll
    for (int t = 0; t < NTW; ++t) acc[t] = (v8f){};

#pragma unroll
    for (int kt = 0; kt < KTILES; ++kt) {
      // run 1: K = kt*32 + koff + {0..7}   (contiguous in c)
      int kk1 = kt * 32 + koff;
      int c1  = (kk1 & (CC - 1)) - koff;            // offset relative to abase
      int t1  = kk1 / CC;                           // known-bits: constant
      int kx1 = t1 % KWC, ky1 = t1 / KWC;
      // run 2: K = kt*32 + koff + {16..23}
      int kk2 = kk1 + 16;
      int c2  = (kk2 & (CC - 1)) - koff;
      int t2  = kk2 / CC;
      int kx2 = t2 % KWC, ky2 = t2 / KWC;

      union { v16h v; v8h h[2]; } a;
      a.h[0] = *(const v8h*)(abase + (ky1 * Win + kx1) * CC + c1);
      a.h[1] = *(const v8h*)(abase + (ky2 * Win + kx2) * CC + c2);

      const h16* brow = bbase + (size_t)kt * Ntiles * 512;
      if (kt + 1 < KTILES)          // stream the next B panel (global_prefetch_b8)
        __builtin_prefetch(bbase + (size_t)(kt + 1) * Ntiles * 512, 0, 1);

#pragma unroll
      for (int t = 0; t < NTW; ++t) {
        v16h bf = *(const v16h*)(brow + (size_t)t * 512);
        acc[t] = __builtin_amdgcn_wmma_f32_16x16x32_f16(
            /*neg_a=*/false, a.v, /*neg_b=*/false, bf,
            /*c_mod=*/(short)0, acc[t], /*reuse_a=*/false, /*reuse_b=*/false);
      }
    }

    int mbase = mt * 16 + ((lane >> 4) << 3);
#pragma unroll
    for (int t = 0; t < NTW; ++t) {
      int ncol = (nt0 + t) * 16 + (lane & 15);
#pragma unroll
      for (int rr = 0; rr < 8; ++rr) {
        float v = acc[t][rr];
        int mm = mbase + rr;
        if (relu) {
          v = fmaxf(v, 0.f);
          float st = (ncol < Nout) ? v : 0.f;
          outH[(size_t)mm * Nstride + ncol] = (h16)st;
          if (ncol < Nout) wmax = fmaxf(wmax, v);
        } else {
          if (ncol < Nout) outF[(size_t)mm * Nstride + ncol] = v;
        }
      }
    }
  }
  if (maxslot) blockMaxAtomic(wmax, maxslot);
}

// fc1 output: quant_relu (ReLU already applied on store) -> padded [B,128] f16
__global__ void k_quant_fc(const h16* __restrict__ r4, const float* __restrict__ maxslot,
                           const int* __restrict__ bwp, h16* __restrict__ h4q, int Btot) {
  int i = blockIdx.x * TPB + threadIdx.x;
  int total = Btot * 128;
  if (i >= total) return;
  int bw = bwp[0];
  float qm = (float)((1 << bw) - 1);
  float s = fmaxf(maxslot[0], 1e-8f) / qm;
  int c = i & 127, b = i >> 7;
  float v = (c < 100) ? (float)r4[(size_t)b * 112 + c] : 0.f;
  float q = fminf(fmaxf(rintf(v / s), 0.f), qm) * s;
  h4q[i] = (h16)q;
}

// ---------------------------------------------------------------------------
extern "C" void kernel_launch(void* const* d_in, const int* in_sizes, int n_in,
                              void* d_out, int out_size, void* d_ws, size_t ws_size,
                              hipStream_t stream) {
  (void)n_in; (void)out_size; (void)ws_size;
  const float* x   = (const float*)d_in[0];
  const float* w1  = (const float*)d_in[1];
  const float* w2  = (const float*)d_in[2];
  const float* w3  = (const float*)d_in[3];
  const float* fw1 = (const float*)d_in[4];
  const float* fw2 = (const float*)d_in[5];
  const int*   bw  = (const int*)d_in[6];
  int B = in_sizes[0] / (3 * 32 * 32);   // 4096

  char* ws = (char*)d_ws;
  size_t off = 0;
  auto alloc = [&](size_t bytes) {
    size_t o = off;
    off += (bytes + 255) & ~(size_t)255;
    return o;
  };
  float* ms  = (float*)(ws + alloc(4 * 4));     // conv1/conv2/conv3/fc1 act maxes
  float* s1  = (float*)(ws + alloc(64 * 4));
  float* s2  = (float*)(ws + alloc(64 * 4));
  float* s3  = (float*)(ws + alloc(64 * 4));
  float* sf1 = (float*)(ws + alloc(100 * 4));
  float* sf2 = (float*)(ws + alloc(131 * 4));
  float* wq1 = (float*)(ws + alloc(64 * 27 * 4));
  h16* pw2 = (h16*)(ws + alloc((size_t)32 * 4 * 512 * 2));  // K=1024, Npad=64
  h16* pw3 = (h16*)(ws + alloc((size_t)18 * 4 * 512 * 2));  // K=576,  Npad=64
  h16* pf1 = (h16*)(ws + alloc((size_t)2 * 7 * 512 * 2));   // K=64,   Npad=112
  h16* pf2 = (h16*)(ws + alloc((size_t)4 * 9 * 512 * 2));   // K=128,  Npad=144
  h16* r1  = (h16*)(ws + alloc((size_t)B * 30 * 30 * 64 * 2));
  h16* h1q = (h16*)(ws + alloc((size_t)B * 15 * 15 * 64 * 2));
  h16* r2  = (h16*)(ws + alloc((size_t)B * 12 * 12 * 64 * 2));
  h16* h2q = (h16*)(ws + alloc((size_t)B * 6 * 6 * 64 * 2));
  h16* r3  = (h16*)(ws + alloc((size_t)B * 2 * 2 * 64 * 2));
  h16* h3q = (h16*)(ws + alloc((size_t)B * 64 * 2));
  h16* r4  = (h16*)(ws + alloc((size_t)B * 112 * 2));
  h16* h4q = (h16*)(ws + alloc((size_t)B * 128 * 2));

  // ---- weight quantization + WMMA-B packing (tiny, once per launch) ----
  k_init<<<1, 32, 0, stream>>>(ms, 4);
  k_amax_perchan<<<64,  TPB, 0, stream>>>(w1,  27,   s1);
  k_amax_perchan<<<64,  TPB, 0, stream>>>(w2,  1024, s2);
  k_amax_perchan<<<64,  TPB, 0, stream>>>(w3,  576,  s3);
  k_amax_perchan<<<100, TPB, 0, stream>>>(fw1, 64,   sf1);
  k_amax_perchan<<<131, TPB, 0, stream>>>(fw2, 100,  sf2);
  k_amax_all<<<1, TPB, 0, stream>>>(sf2, 131);            // fw2 is per-tensor
  k_quant_conv1<<<(64 * 27 + TPB - 1) / TPB, TPB, 0, stream>>>(w1, s1, bw, wq1, 64 * 27, 27);
  k_pack_w<<<(32 * 4 * 512) / TPB, TPB, 0, stream>>>(w2,  s2,  bw, pw2, 64,  64,  4, 4, 32, 4);
  k_pack_w<<<(18 * 4 * 512) / TPB, TPB, 0, stream>>>(w3,  s3,  bw, pw3, 64,  64,  3, 3, 18, 4);
  k_pack_w<<<(2 * 7 * 512)  / TPB, TPB, 0, stream>>>(fw1, sf1, bw, pf1, 100, 64,  1, 1, 2,  7);
  k_pack_w<<<(4 * 9 * 512)  / TPB, TPB, 0, stream>>>(fw2, sf2, bw, pf2, 131, 100, 1, 1, 4,  9);

  // ---- conv1 (scalar, fused relu+max) -> quant+pool ----
  { size_t tot = (size_t)B * 30 * 30 * 64;
    k_conv1<<<(unsigned)((tot + TPB - 1) / TPB), TPB, 0, stream>>>(x, wq1, r1, ms + 0, B); }
  { size_t tot = (size_t)B * 15 * 15 * 64;
    k_quant_pool<<<(unsigned)((tot + TPB - 1) / TPB), TPB, 0, stream>>>(r1, ms + 0, bw, h1q, B, 15, 15, 64); }

  // ---- conv2: M=B*144, N=64, K=1024 (WMMA, 16x64/wave) -> quant+pool ----
  { int Mt = B * 144 / 16;
    int waves = Mt;                                   // NTW=4 covers all N
    k_gemm_wmma<64, 4, 1, 32, 4><<<(waves * 32 + TPB - 1) / TPB, TPB, 0, stream>>>(
        h1q, pw2, r2, nullptr, ms + 1, 15, 15, 12, 12, 4, Mt, 64, 64, 1); }
  { size_t tot = (size_t)B * 6 * 6 * 64;
    k_quant_pool<<<(unsigned)((tot + TPB - 1) / TPB), TPB, 0, stream>>>(r2, ms + 1, bw, h2q, B, 6, 6, 64); }

  // ---- conv3 (stride 2): M=B*4, N=64, K=576 (WMMA, 16x64/wave) -> quant+pool ----
  { int Mt = B * 4 / 16;
    int waves = Mt;
    k_gemm_wmma<64, 3, 2, 18, 4><<<(waves * 32 + TPB - 1) / TPB, TPB, 0, stream>>>(
        h2q, pw3, r3, nullptr, ms + 2, 6, 6, 2, 2, 4, Mt, 64, 64, 1); }
  { size_t tot = (size_t)B * 64;
    k_quant_pool<<<(unsigned)((tot + TPB - 1) / TPB), TPB, 0, stream>>>(r3, ms + 2, bw, h3q, B, 1, 1, 64); }

  // ---- fc1: M=B, N=100 (pad 112), K=64 (WMMA) -> quant ----
  { int Mt = B / 16, Nt = 7;
    int waves = Mt * Nt;
    k_gemm_wmma<64, 1, 1, 2, 1><<<(waves * 32 + TPB - 1) / TPB, TPB, 0, stream>>>(
        h3q, pf1, r4, nullptr, ms + 3, 1, 1, 1, 1, Nt, Mt, 100, 112, 1); }
  { int tot = B * 128;
    k_quant_fc<<<(tot + TPB - 1) / TPB, TPB, 0, stream>>>(r4, ms + 3, bw, h4q, B); }

  // ---- fc2: M=B, N=131 (pad 144), K=100 (pad 128) (WMMA) -> d_out f32 ----
  { int Mt = B / 16, Nt = 9;
    int waves = Mt * Nt;
    k_gemm_wmma<128, 1, 1, 4, 1><<<(waves * 32 + TPB - 1) / TPB, TPB, 0, stream>>>(
        h4q, pf2, nullptr, (float*)d_out, nullptr, 1, 1, 1, 1, Nt, Mt, 131, 131, 0); }
}
